// SelfAttention_Transpose_2319282340249
// MI455X (gfx1250) — compile-verified, matrix operand order checked by reading, not run on hardware
//
#include <hip/hip_runtime.h>
#include <stdint.h>

// ---------------------------------------------------------------------------
// Problem constants (match reference)
// ---------------------------------------------------------------------------
#define DD 1024   // rnn_size
#define HH 64     // pooling hidden
#define BB 16     // batch
#define CC 1024   // context length

// GEMM tiling
#define BM 128
#define BN 128
#define BK 32
#define LDT 40    // padded LDS row stride in bf16 elements

typedef __attribute__((ext_vector_type(16))) __bf16        v16bf;
typedef __attribute__((ext_vector_type(8)))  float         v8f;
typedef __attribute__((ext_vector_type(8)))  unsigned int  v8u;

// ---------------------------------------------------------------------------
// CDNA5 async LDS copy support (guarded; sync fallback)
// Builtin prototype (from compiler diagnostic):
//   void __builtin_amdgcn_global_load_async_to_lds_b64(
//       __attribute__((vector_size(8))) int AS1* gaddr,
//       __attribute__((vector_size(8))) int AS3* ldsaddr,
//       imm int offset, imm int cpol);
// ---------------------------------------------------------------------------
#if __has_builtin(__builtin_amdgcn_global_load_async_to_lds_b64)
#define ASYNC_LDS 1
typedef __attribute__((__vector_size__(8))) int i32x2;
typedef __attribute__((address_space(1))) i32x2* gv2_t;
typedef __attribute__((address_space(3))) i32x2* lv2_t;
#define AS_G64(p) ((gv2_t)(uintptr_t)(p))
#define AS_L64(p) ((lv2_t)(unsigned int)(uintptr_t)(p))
#else
#define ASYNC_LDS 0
#endif

__device__ __forceinline__ void wait_async_zero() {
#if ASYNC_LDS
#if __has_builtin(__builtin_amdgcn_s_wait_asynccnt)
  __builtin_amdgcn_s_wait_asynccnt(0);
#else
  asm volatile("s_wait_asynccnt 0x0" ::: "memory");
#endif
#endif
}

// ---------------------------------------------------------------------------
// bf16 helpers (manual RNE; bf16 kept as u16 bit patterns)
// ---------------------------------------------------------------------------
__device__ __forceinline__ unsigned short f2bf(float f) {
  union { float f; unsigned u; } x; x.f = f;
  unsigned r = x.u + 0x7FFFu + ((x.u >> 16) & 1u);
  return (unsigned short)(r >> 16);
}

// ---------------------------------------------------------------------------
// Elementwise fp32 -> bf16 (vec4)
// ---------------------------------------------------------------------------
__global__ __launch_bounds__(256)
void cvt_bf16x4(const float* __restrict__ s, unsigned short* __restrict__ d, long n)
{
  const long i = ((long)blockIdx.x * 256 + threadIdx.x) * 4;
  if (i >= n) return;
  const float4 v = *(const float4*)(s + i);
  uint2 pk;
  pk.x = (unsigned)f2bf(v.x) | ((unsigned)f2bf(v.y) << 16);
  pk.y = (unsigned)f2bf(v.z) | ((unsigned)f2bf(v.w) << 16);
  *(uint2*)(d + i) = pk;
}

// ---------------------------------------------------------------------------
// W [K=D][N=D] fp32  ->  WT [N][K] bf16 (32x32 LDS tile transpose)
// Grid: (D/32, D/32), block 256.
// ---------------------------------------------------------------------------
__global__ __launch_bounds__(256)
void transpose_cvt_bf16(const float* __restrict__ W, unsigned short* __restrict__ WT)
{
  __shared__ float tile[32][33];
  const int tx = threadIdx.x & 31;
  const int ty = threadIdx.x >> 5;      // 0..7
  const int k0 = blockIdx.x * 32;
  const int n0 = blockIdx.y * 32;
  #pragma unroll
  for (int r = 0; r < 4; ++r) {
    const int row = ty + r * 8;
    tile[row][tx] = W[(long)(k0 + row) * DD + n0 + tx];
  }
  __syncthreads();
  #pragma unroll
  for (int r = 0; r < 4; ++r) {
    const int row = ty + r * 8;
    WT[(long)(n0 + row) * DD + k0 + tx] = f2bf(tile[tx][row]);
  }
}

// ---------------------------------------------------------------------------
// Single-layout WMMA GEMM (NT):  C[m,n] = act( sum_k A[m,k]*B[n,k] + bias[n] )
//   A: [M,K] bf16 row-major, B: [N,K] bf16 row-major.
//   Double-buffered LDS staging via GLOBAL_LOAD_ASYNC_TO_LDS (ASYNCcnt) when
//   available; synchronous 8-byte copies otherwise.
//   Outputs: C32 fp32 and/or C16 bf16 (nullable). c_transpose stores C16 as
//   per-batch [N][C] (used to produce v^T for the attn^T@v GEMM).
//   Block 256 = 8 waves; each wave owns a 16x128 strip (8 wmma / K-step).
//   Grid: (N/128, M/128, batches)
// ---------------------------------------------------------------------------
__global__ __launch_bounds__(256)
void gemm_bf16_nt(const unsigned short* __restrict__ A, long lda, long strideA,
                  const unsigned short* __restrict__ B, long ldb, long strideB,
                  const float* __restrict__ bias, int relu,
                  float* __restrict__ C32, unsigned short* __restrict__ C16,
                  int c_transpose, long ldc, long strideC, int K)
{
  __shared__ unsigned short Al[2 * BM * LDT];
  __shared__ unsigned short Bl[2 * BN * LDT];

  const int  t    = threadIdx.x;
  const int  wave = t >> 5;
  const int  lane = t & 31;
  const int  half = lane >> 4;      // K-group select (ISA 16-bit layout)
  const int  lm   = lane & 15;      // row(A) / col(B) within 16
  const long mb   = (long)blockIdx.y * BM;
  const long nb   = (long)blockIdx.x * BN;
  const long z    = blockIdx.z;

  const unsigned short* Ag0 = A + strideA * z + mb * lda;
  const unsigned short* Bg0 = B + strideB * z + nb * ldb;

  // stage BMxBK (A) and BNxBK (B) tiles into LDS buffer `buf`
  auto stage = [&](int buf, int k0) {
    unsigned short* Ad = &Al[buf * BM * LDT];
    unsigned short* Bd = &Bl[buf * BN * LDT];
    #pragma unroll
    for (int it = 0; it < 4; ++it) {              // 1024 chunks / 256 thr
      const int c = t + it * 256;
      const int r = c >> 3, p = (c & 7) * 4;      // row, 4-elem part
#if ASYNC_LDS
      __builtin_amdgcn_global_load_async_to_lds_b64(
          AS_G64(Ag0 + (long)r * lda + k0 + p), AS_L64(Ad + r * LDT + p), 0, 0);
      __builtin_amdgcn_global_load_async_to_lds_b64(
          AS_G64(Bg0 + (long)r * ldb + k0 + p), AS_L64(Bd + r * LDT + p), 0, 0);
#else
      *(uint2*)(Ad + r * LDT + p) = *(const uint2*)(Ag0 + (long)r * lda + k0 + p);
      *(uint2*)(Bd + r * LDT + p) = *(const uint2*)(Bg0 + (long)r * ldb + k0 + p);
#endif
    }
  };

  v8f acc[8] = {};
  int buf = 0;
  stage(0, 0);

  for (int k0 = 0; k0 < K; k0 += BK) {
    wait_async_zero();          // my async chunks arrived
    __syncthreads();            // everyone's chunks arrived; prev compute done
    if (k0 + BK < K) stage(buf ^ 1, k0 + BK);

    const unsigned short* As = &Al[buf * BM * LDT];
    const unsigned short* Bs = &Bl[buf * BN * LDT];

    // A fragment (ISA 16-bit A layout: K pairs per VGPR, half*8 K offset)
    const int arow = wave * 16 + lm;
    v8u au;
    #pragma unroll
    for (int j = 0; j < 8; ++j) {
      const int kk = (j < 4 ? j * 2 : 16 + (j - 4) * 2) + half * 8;
      au[j] = *(const unsigned int*)&As[arow * LDT + kk];
    }
    const v16bf af = __builtin_bit_cast(v16bf, au);

    #pragma unroll
    for (int nt = 0; nt < 8; ++nt) {
      v8u bu;
      #pragma unroll
      for (int j = 0; j < 8; ++j) {
        const int kk = (j < 4 ? j * 2 : 16 + (j - 4) * 2) + half * 8;
        bu[j] = *(const unsigned int*)&Bs[(nt * 16 + lm) * LDT + kk];
      }
      const v16bf bfr = __builtin_bit_cast(v16bf, bu);
      acc[nt] = __builtin_amdgcn_wmma_f32_16x16x32_bf16(
          false, af, false, bfr, (short)0, acc[nt], false, false);
    }
    buf ^= 1;
  }

  // ---- epilogue: bias + relu, scatter per ISA C/D layout ------------------
  const long coff = strideC * z;
  #pragma unroll
  for (int nt = 0; nt < 8; ++nt) {
    #pragma unroll
    for (int r = 0; r < 8; ++r) {
      const long m = mb + wave * 16 + half * 8 + r;
      const long n = nb + nt * 16 + lm;
      float v = acc[nt][r];
      if (bias) v += bias[n];
      if (relu) v = fmaxf(v, 0.f);
      if (C32) C32[coff + m * ldc + n] = v;
      if (C16) {
        if (c_transpose) {
          // m = b*C + i  ->  store at [b][n][i]  (per-batch [N][C])
          const long bb = m >> 10;            // /CC
          const long ii = m & (CC - 1);
          C16[bb * (long)CC * DD + n * CC + ii] = f2bf(v);
        } else {
          C16[coff + m * ldc + n] = f2bf(v);
        }
      }
    }
  }
}

// ---------------------------------------------------------------------------
// Softmax over axis=1 (columns) of energy[b][i][j]; writes TRANSPOSED bf16
// attn^T[b][j][i] so the next GEMM reads it as A:[M=j][K=i] row-major.
// Block 256 = 32 cols x 8 row-strips. Grid: (C/32, B)
// ---------------------------------------------------------------------------
__global__ __launch_bounds__(256)
void col_softmax_t(const float* __restrict__ E, unsigned short* __restrict__ OT)
{
  const int b  = blockIdx.y;
  const int tj = threadIdx.x & 31;
  const int ti = threadIdx.x >> 5;      // 0..7
  const int j  = blockIdx.x * 32 + tj;
  __shared__ float red[8][32];

  const float* Eb = E + (long)b * CC * CC;

  float m = -1e30f;
  for (int i = ti; i < CC; i += 8) m = fmaxf(m, Eb[(long)i * CC + j]);
  red[ti][tj] = m;
  __syncthreads();
  if (ti == 0) {
    float mm = red[0][tj];
    #pragma unroll
    for (int r = 1; r < 8; ++r) mm = fmaxf(mm, red[r][tj]);
    red[0][tj] = mm;
  }
  __syncthreads();
  const float cm = red[0][tj];
  __syncthreads();

  float s = 0.f;
  for (int i = ti; i < CC; i += 8) s += __expf(Eb[(long)i * CC + j] - cm);
  red[ti][tj] = s;
  __syncthreads();
  if (ti == 0) {
    float ss = red[0][tj];
    #pragma unroll
    for (int r = 1; r < 8; ++r) ss += red[r][tj];
    red[0][tj] = ss;
  }
  __syncthreads();
  const float inv = 1.0f / red[0][tj];

  unsigned short* Ob = OT + (long)b * CC * CC + (long)j * CC;  // row j of attn^T
  for (int i = ti; i < CC; i += 8)
    Ob[i] = f2bf(__expf(Eb[(long)i * CC + j] - cm) * inv);
}

// ---------------------------------------------------------------------------
// MLP pooling scores: score[row] = relu(x[row,:] @ W1 + b1) @ W2 + b2
// ---------------------------------------------------------------------------
__global__ __launch_bounds__(256)
void mlp_score(const float* __restrict__ x, const float* __restrict__ W1,
               const float* __restrict__ b1, const float* __restrict__ W2,
               const float* __restrict__ b2, float* __restrict__ score)
{
  const long row = blockIdx.x;
  const int  t   = threadIdx.x;
  __shared__ float xs[DD];
  __shared__ float hred[HH];

  for (int d = t; d < DD; d += 256) xs[d] = x[row * DD + d];
  __syncthreads();

  if (t < HH) {
    float a = b1[t];
    for (int d = 0; d < DD; ++d) a = fmaf(xs[d], W1[(long)d * HH + t], a);
    a = fmaxf(a, 0.f);
    hred[t] = a * W2[t];
  }
  __syncthreads();
  if (t == 0) {
    float s = b2[0];
    #pragma unroll
    for (int h = 0; h < HH; ++h) s += hred[h];
    score[row] = s;
  }
}

// ---------------------------------------------------------------------------
// Row softmax over C for pooling weights. One block per batch.
// ---------------------------------------------------------------------------
__global__ __launch_bounds__(256)
void row_softmax(const float* __restrict__ s, float* __restrict__ w)
{
  const int b = blockIdx.x, t = threadIdx.x;
  __shared__ float red[256];
  const float* sb = s + (long)b * CC;

  float m = -1e30f;
  for (int j = t; j < CC; j += 256) m = fmaxf(m, sb[j]);
  red[t] = m; __syncthreads();
  for (int o = 128; o > 0; o >>= 1) {
    if (t < o) red[t] = fmaxf(red[t], red[t + o]);
    __syncthreads();
  }
  m = red[0]; __syncthreads();

  float a = 0.f;
  for (int j = t; j < CC; j += 256) a += __expf(sb[j] - m);
  red[t] = a; __syncthreads();
  for (int o = 128; o > 0; o >>= 1) {
    if (t < o) red[t] += red[t + o];
    __syncthreads();
  }
  const float inv = 1.0f / red[0];

  for (int j = t; j < CC; j += 256) w[(long)b * CC + j] = __expf(sb[j] - m) * inv;
}

// ---------------------------------------------------------------------------
// Final: out[b,d] = sum_j (gamma*ao[b,j,d] + x[b,j,d]) * w[b,j]
// Grid: (D/256, B)
// ---------------------------------------------------------------------------
__global__ __launch_bounds__(256)
void finalize_pool(const float* __restrict__ x, const float* __restrict__ ao,
                   const float* __restrict__ w, const float* __restrict__ gamma,
                   float* __restrict__ out)
{
  const int b = blockIdx.y;
  const int d = blockIdx.x * 256 + threadIdx.x;
  __shared__ float ws_[CC];
  for (int j = threadIdx.x; j < CC; j += 256) ws_[j] = w[(long)b * CC + j];
  __syncthreads();

  const float g  = gamma[0];
  const float* xb = x  + (long)b * CC * DD + d;
  const float* ab = ao + (long)b * CC * DD + d;
  float acc = 0.f;
  for (int j = 0; j < CC; ++j)
    acc = fmaf(g * ab[(long)j * DD] + xb[(long)j * DD], ws_[j], acc);
  out[(long)b * DD + d] = acc;
}

// ---------------------------------------------------------------------------
// Host launcher
// ---------------------------------------------------------------------------
extern "C" void kernel_launch(void* const* d_in, const int* in_sizes, int n_in,
                              void* d_out, int out_size, void* d_ws, size_t ws_size,
                              hipStream_t stream)
{
  (void)in_sizes; (void)n_in; (void)out_size; (void)ws_size;

  const float* x    = (const float*)d_in[0];
  const float* Wf   = (const float*)d_in[1];
  const float* bf_  = (const float*)d_in[2];
  const float* Wg   = (const float*)d_in[3];
  const float* bg   = (const float*)d_in[4];
  const float* Wx   = (const float*)d_in[5];
  const float* bx   = (const float*)d_in[6];
  const float* W1   = (const float*)d_in[7];
  const float* b1   = (const float*)d_in[8];
  const float* W2   = (const float*)d_in[9];
  const float* b2   = (const float*)d_in[10];
  const float* gam  = (const float*)d_in[11];
  float* out = (float*)d_out;

  // workspace carve-out
  char* ws = (char*)d_ws;
  size_t off = 0;
  auto carve = [&](size_t bytes) -> char* {
    char* p = ws + off;
    off += (bytes + 255) & ~(size_t)255;
    return p;
  };
  unsigned short* xb     = (unsigned short*)carve((size_t)BB * CC * DD * 2);
  unsigned short* WfT    = (unsigned short*)carve((size_t)DD * DD * 2);
  unsigned short* WgT    = (unsigned short*)carve((size_t)DD * DD * 2);
  unsigned short* WxT    = (unsigned short*)carve((size_t)DD * DD * 2);
  unsigned short* qb     = (unsigned short*)carve((size_t)BB * CC * DD * 2);
  unsigned short* kb     = (unsigned short*)carve((size_t)BB * CC * DD * 2);
  unsigned short* vTb    = (unsigned short*)carve((size_t)BB * CC * DD * 2);
  float*          energy = (float*)         carve((size_t)BB * CC * CC * 4);
  unsigned short* attnT  = (unsigned short*)carve((size_t)BB * CC * CC * 2);
  float*          ao     = (float*)         carve((size_t)BB * CC * DD * 4);
  float*          score  = (float*)         carve((size_t)BB * CC * 4);
  float*          w      = (float*)         carve((size_t)BB * CC * 4);

  const dim3 blk(256);

  // 0) precision pass: x -> bf16; W -> transposed bf16 ([N][K])
  const long nx = (long)BB * CC * DD;
  cvt_bf16x4<<<dim3((unsigned)(nx / 4 / 256)), blk, 0, stream>>>(x, xb, nx);
  const dim3 gtw(DD / 32, DD / 32);
  transpose_cvt_bf16<<<gtw, blk, 0, stream>>>(Wf, WfT);
  transpose_cvt_bf16<<<gtw, blk, 0, stream>>>(Wg, WgT);
  transpose_cvt_bf16<<<gtw, blk, 0, stream>>>(Wx, WxT);

  // 1) projections: q/k = relu(x@W+b) -> bf16 [B*C, D]; v stored transposed
  const dim3 gproj(DD / BN, (BB * CC) / BM, 1);
  gemm_bf16_nt<<<gproj, blk, 0, stream>>>(xb, DD, 0, WfT, DD, 0, bf_, 1,
                                          nullptr, qb, 0, DD, 0, DD);
  gemm_bf16_nt<<<gproj, blk, 0, stream>>>(xb, DD, 0, WgT, DD, 0, bg, 1,
                                          nullptr, kb, 0, DD, 0, DD);
  gemm_bf16_nt<<<gproj, blk, 0, stream>>>(xb, DD, 0, WxT, DD, 0, bx, 1,
                                          nullptr, vTb, 1, DD, 0, DD);

  // 2) energy[b] = q[b] @ k[b]^T  (A=[C,D], B=[C,D] as [N,K])
  const dim3 gen(CC / BN, CC / BM, BB);
  gemm_bf16_nt<<<gen, blk, 0, stream>>>(qb, DD, (long)CC * DD,
                                        kb, DD, (long)CC * DD,
                                        nullptr, 0, energy, nullptr, 0,
                                        CC, (long)CC * CC, DD);

  // 3) attn^T = softmax over axis=1, stored [b][j][i] bf16
  col_softmax_t<<<dim3(CC / 32, BB), blk, 0, stream>>>(energy, attnT);

  // 4) ao[b] = attn[b]^T @ v[b]  (A=attn^T [j][i], B=v^T [d][i] as [N,K])
  const dim3 gav(DD / BN, CC / BM, BB);
  gemm_bf16_nt<<<gav, blk, 0, stream>>>(attnT, CC, (long)CC * CC,
                                        vTb,   CC, (long)CC * DD,
                                        nullptr, 0, ao, nullptr, 0,
                                        DD, (long)CC * DD, CC);

  // 5) pooling scores + softmax + fused gamma-residual weighted sum
  mlp_score<<<dim3(BB * CC), blk, 0, stream>>>(x, W1, b1, W2, b2, score);
  row_softmax<<<dim3(BB), blk, 0, stream>>>(score, w);
  finalize_pool<<<dim3(DD / 256, BB), blk, 0, stream>>>(x, ao, w, gam, out);
}
